// LSTM_52596169507493
// MI455X (gfx1250) — matrix operand order repaired and checked by reading, not compiled, not run on hardware
//
#include <hip/hip_runtime.h>
#include <hip/hip_bf16.h>
#include <stdint.h>

// ---------------------------------------------------------------------------
// LSTM cell + logits projection for MI455X (gfx1250, wave32, WMMA + TDM).
//  - bf16 WMMA (f32 accum); W_logits bf16 = 103MB -> resident in 192MB L2.
//  - Weights pre-transposed to K-major so every GEMM tile is a straight 2D
//    tile copy -> staged by the Tensor Data Mover (tensor_load_to_lds) with
//    LDS row padding done by the TDM pad fields; double-buffered ping/pong.
// ---------------------------------------------------------------------------

#define B_ROWS 4096
#define EMB    300
#define EMBP   320            // EMB padded to multiple of 32
#define HID    1024
#define KC     (EMBP + HID)   // 1344
#define VOCAB  50257
#define NLP    50304          // VOCAB padded to multiple of 128

typedef __bf16 bf16_t;
typedef bf16_t v16bf __attribute__((ext_vector_type(16)));
typedef float  v8f   __attribute__((ext_vector_type(8)));
typedef unsigned int u32x4 __attribute__((ext_vector_type(4)));
typedef int          i32x8 __attribute__((ext_vector_type(8)));
typedef int          i32x4 __attribute__((ext_vector_type(4)));

#if __has_builtin(__builtin_amdgcn_tensor_load_to_lds)
#define HAS_TDM 1
#else
#define HAS_TDM 0
#endif

__device__ __forceinline__ unsigned short f2bf(float f) {
  unsigned int u = __float_as_uint(f);
  u += 0x7FFFu + ((u >> 16) & 1u);      // round-to-nearest-even
  return (unsigned short)(u >> 16);
}

union FragU { uint4 q[2]; v16bf v; };

#if HAS_TDM
// 2D bf16 tile load, global -> LDS, with TDM pad producing a 40-ushort
// (80-byte) LDS row pitch for 32-element rows: pad_interval=16 DW, pad=4 DW.
// This toolchain exposes the 6-arg builtin:
//   (u32x4 group0, i32x8 group1, i32x4 group2, i32x4 group3, i32x8 extra, i32 cpol)
__device__ __forceinline__ void tdm_load_2d(const void* gptr, unsigned lds_byte,
                                            unsigned rows, unsigned stride_elems) {
  unsigned long long ga = (unsigned long long)(uintptr_t)gptr;
  u32x4 g0;
  g0[0] = 1u;                                  // count=1, user mode, no gather
  g0[1] = lds_byte;                            // LDS byte address
  g0[2] = (unsigned)ga;                        // global_addr[31:0]
  g0[3] = ((unsigned)(ga >> 32) & 0x01FFFFFFu) | (2u << 30);  // addr hi | type=2
  i32x8 g1;
  g1[0] = (1 << 16) | (1 << 20) | (3 << 22) | (3 << 25); // 2B elems, pad_en, 16DW/4DW
  g1[1] = (int)((stride_elems & 0xFFFFu) << 16);         // tensor_dim0 lo16
  g1[2] = (int)(((stride_elems >> 16) & 0xFFFFu) | ((rows & 0xFFFFu) << 16)); // d0 hi | d1 lo
  g1[3] = (int)(((rows >> 16) & 0xFFFFu) | (32u << 16)); // d1 hi | tile_dim0 = 32
  g1[4] = (int)(rows & 0xFFFFu);                         // tile_dim1 = rows, tile_dim2 = 0
  g1[5] = (int)stride_elems;                             // tensor_dim0_stride lo32
  g1[6] = 0;                                             // stride hi | dim1_stride lo
  g1[7] = 0;
  i32x4 z4 = {0, 0, 0, 0};
  i32x8 z8 = {0, 0, 0, 0, 0, 0, 0, 0};
  __builtin_amdgcn_tensor_load_to_lds(g0, g1, z4, z4, z8, 0);
}
#endif

// ---------------------------------------------------------------------------
// Packing kernels
// ---------------------------------------------------------------------------

__global__ __launch_bounds__(256) void pack_A(const int* __restrict__ Xidx,
                                              const float* __restrict__ H,
                                              const float* __restrict__ emb,
                                              unsigned short* __restrict__ Acat) {
  int idx = blockIdx.x * 256 + threadIdx.x;   // over 4096*1344
  int m = idx / KC, c = idx % KC;
  float v;
  if (c < EMB)       v = emb[(size_t)Xidx[m] * EMB + c];
  else if (c < EMBP) v = 0.0f;
  else               v = H[(size_t)m * HID + (c - EMBP)];
  Acat[idx] = f2bf(v);
}

// Gate weights: stack [Wx ; 0-pad ; Wh] and transpose -> WcatT[g][n][k] bf16.
__global__ __launch_bounds__(256) void packT_Wcat(const float* __restrict__ Wfx, const float* __restrict__ Wfh,
                                                  const float* __restrict__ Wix, const float* __restrict__ Wih,
                                                  const float* __restrict__ Wgx, const float* __restrict__ Wgh,
                                                  const float* __restrict__ Wox, const float* __restrict__ Woh,
                                                  unsigned short* __restrict__ WcatT) {
  __shared__ unsigned short t[32][33];
  const int g  = blockIdx.z;
  const float* Wx = (g == 0) ? Wfx : (g == 1) ? Wix : (g == 2) ? Wgx : Wox;
  const float* Wh = (g == 0) ? Wfh : (g == 1) ? Wih : (g == 2) ? Wgh : Woh;
  const int k0 = blockIdx.x * 32, n0 = blockIdx.y * 32;
  const int tx = threadIdx.x & 31, ty = threadIdx.x >> 5;   // ty 0..7
#pragma unroll
  for (int j = 0; j < 4; ++j) {
    int kr = ty + j * 8;
    int k = k0 + kr, n = n0 + tx;
    float v;
    if (k < EMB)       v = Wx[(size_t)k * HID + n];
    else if (k < EMBP) v = 0.0f;
    else               v = Wh[(size_t)(k - EMBP) * HID + n];
    t[kr][tx] = f2bf(v);
  }
  __syncthreads();
#pragma unroll
  for (int j = 0; j < 4; ++j) {
    int nr = ty + j * 8;
    WcatT[((size_t)g * HID + n0 + nr) * KC + k0 + tx] = t[tx][nr];
  }
}

// W_logits: transpose + pad N to NLP -> WlogT[n][k] bf16.
__global__ __launch_bounds__(256) void packT_Wlog(const float* __restrict__ Wlog,
                                                  unsigned short* __restrict__ WlogT) {
  __shared__ unsigned short t[32][33];
  const int k0 = blockIdx.x * 32, n0 = blockIdx.y * 32;
  const int tx = threadIdx.x & 31, ty = threadIdx.x >> 5;
#pragma unroll
  for (int j = 0; j < 4; ++j) {
    int kr = ty + j * 8;
    int n = n0 + tx;
    float v = (n < VOCAB) ? Wlog[(size_t)(k0 + kr) * VOCAB + n] : 0.0f;
    t[kr][tx] = f2bf(v);
  }
  __syncthreads();
#pragma unroll
  for (int j = 0; j < 4; ++j) {
    int nr = ty + j * 8;
    WlogT[(size_t)(n0 + nr) * HID + k0 + tx] = t[tx][nr];
  }
}

__global__ __launch_bounds__(256) void pack_bsum(const float* __restrict__ bfx, const float* __restrict__ bfh,
                                                 const float* __restrict__ bix, const float* __restrict__ bih,
                                                 const float* __restrict__ bgx, const float* __restrict__ bgh,
                                                 const float* __restrict__ box, const float* __restrict__ boh,
                                                 float* __restrict__ bsum) {
  int idx = blockIdx.x * 256 + threadIdx.x;   // over 4*1024
  int g = idx >> 10, n = idx & 1023;
  const float* bx = (g == 0) ? bfx : (g == 1) ? bix : (g == 2) ? bgx : box;
  const float* bh = (g == 0) ? bfh : (g == 1) ? bih : (g == 2) ? bgh : boh;
  bsum[idx] = bx[n] + bh[n];
}

// Fallback cooperative straight-copy staging (used only if TDM builtin absent):
// copies a [rows x 32] bf16 tile (row pitch stride_elems) into LDS pitch 40.
__device__ __forceinline__ void copy_tile(const unsigned short* __restrict__ src,
                                          unsigned short* __restrict__ dst,
                                          int rows, int stride_elems, int tid) {
  for (int e = tid; e < rows * 2; e += 256) {     // 2 chunks of 16 bf16 per row
    int r = e >> 1, s = (e & 1) * 16;
    const uint4* g = (const uint4*)(src + (size_t)r * stride_elems + s);
    uint4 q0 = g[0], q1 = g[1];
    uint4* d = (uint4*)(dst + r * 40 + s);
    d[0] = q0; d[1] = q1;
  }
}

// ---------------------------------------------------------------------------
// Fused 4-gate GEMM + LSTM epilogue.  Block = 8 waves (4M x 2N), tile 128x64.
// ---------------------------------------------------------------------------

__global__ __launch_bounds__(256) void gates_lstm(const unsigned short* __restrict__ Acat,
                                                  const unsigned short* __restrict__ WcatT,
                                                  const float* __restrict__ bsum,
                                                  const float* __restrict__ C,
                                                  unsigned short* __restrict__ HnewBF) {
  __shared__ __align__(16) unsigned short lA[2][128 * 40];     // ping/pong [m][k]
  __shared__ __align__(16) unsigned short lB[2][4][64 * 40];   // ping/pong [g][n][k]

  const int tid  = threadIdx.x;
  const int lane = tid & 31;
  const int w    = tid >> 5;
  const int wm   = w & 3;
  const int wn   = w >> 2;
  const int bm   = blockIdx.y * 128;
  const int bn   = blockIdx.x * 64;
  const int NT   = KC / 32;                  // 42 K-tiles

  v8f acc[4][2][2];
  const v8f vzero = {0.f, 0.f, 0.f, 0.f, 0.f, 0.f, 0.f, 0.f};
#pragma unroll
  for (int g = 0; g < 4; ++g)
#pragma unroll
    for (int mi = 0; mi < 2; ++mi)
#pragma unroll
      for (int ni = 0; ni < 2; ++ni) acc[g][mi][ni] = vzero;

#if HAS_TDM
  if (w == 0) {   // tile 0 into buffer 0 : 1 A-tile + 4 B-tiles = 5 TDM ops
    tdm_load_2d(Acat + (size_t)bm * KC, (unsigned)(size_t)(void*)&lA[0][0], 128, KC);
#pragma unroll
    for (int g = 0; g < 4; ++g)
      tdm_load_2d(WcatT + ((size_t)g * HID + bn) * KC,
                  (unsigned)(size_t)(void*)&lB[0][g][0], 64, KC);
  }
#else
  copy_tile(Acat + (size_t)bm * KC, &lA[0][0], 128, KC, tid);
  for (int g = 0; g < 4; ++g)
    copy_tile(WcatT + ((size_t)g * HID + bn) * KC, &lB[0][g][0], 64, KC, tid);
#endif

  for (int kt = 0; kt < NT; ++kt) {
    const int cur = kt & 1, nxt = cur ^ 1;
#if HAS_TDM
    if (w == 0) {
      if (kt + 1 < NT) {
        const int kk = (kt + 1) * 32;
        tdm_load_2d(Acat + (size_t)bm * KC + kk, (unsigned)(size_t)(void*)&lA[nxt][0], 128, KC);
#pragma unroll
        for (int g = 0; g < 4; ++g)
          tdm_load_2d(WcatT + ((size_t)g * HID + bn) * KC + kk,
                      (unsigned)(size_t)(void*)&lB[nxt][g][0], 64, KC);
        __builtin_amdgcn_s_wait_tensorcnt((short)5);   // current tile landed
      } else {
        __builtin_amdgcn_s_wait_tensorcnt((short)0);
      }
    }
#else
    if (kt + 1 < NT) {
      const int kk = (kt + 1) * 32;
      copy_tile(Acat + (size_t)bm * KC + kk, &lA[nxt][0], 128, KC, tid);
      for (int g = 0; g < 4; ++g)
        copy_tile(WcatT + ((size_t)g * HID + bn) * KC + kk, &lB[nxt][g][0], 64, KC, tid);
    }
#endif
    __syncthreads();

    // fragment loads (wave32 WMMA layouts) + 16 WMMAs
    v16bf afr[2];
    const int ar0  = wm * 32 + (lane & 15);
    const int aoff = (lane < 16) ? 0 : 8;      // ushort units
#pragma unroll
    for (int mi = 0; mi < 2; ++mi) {
      FragU au;
      au.q[0] = *(const uint4*)(&lA[cur][0] + (ar0 + mi * 16) * 40 + aoff);
      au.q[1] = *(const uint4*)(&lA[cur][0] + (ar0 + mi * 16) * 40 + aoff + 16);
      afr[mi] = au.v;
    }
    const int bc0  = wn * 32 + (lane & 15);
    const int koff = (lane < 16) ? 0 : 8;
#pragma unroll
    for (int g = 0; g < 4; ++g) {
#pragma unroll
      for (int ni = 0; ni < 2; ++ni) {
        FragU bu;
        bu.q[0] = *(const uint4*)(&lB[cur][g][0] + (bc0 + ni * 16) * 40 + koff);
        bu.q[1] = *(const uint4*)(&lB[cur][g][0] + (bc0 + ni * 16) * 40 + koff + 16);
#pragma unroll
        for (int mi = 0; mi < 2; ++mi)
          acc[g][mi][ni] = __builtin_amdgcn_wmma_f32_16x16x32_bf16(
              false, afr[mi], false, bu.v, (short)0, acc[g][mi][ni], false, false);
      }
    }
    __syncthreads();   // all readers done before buffer 'cur' is overwritten
  }

  // LSTM pointwise epilogue
#pragma unroll
  for (int mi = 0; mi < 2; ++mi) {
#pragma unroll
    for (int ni = 0; ni < 2; ++ni) {
      const int m0 = bm + wm * 32 + mi * 16 + ((lane < 16) ? 0 : 8);
      const int n0 = bn + wn * 32 + ni * 16 + (lane & 15);
      const float bf_ = bsum[0 * HID + n0];
      const float bi_ = bsum[1 * HID + n0];
      const float bg_ = bsum[2 * HID + n0];
      const float bo_ = bsum[3 * HID + n0];
#pragma unroll
      for (int r = 0; r < 8; ++r) {
        const int m = m0 + r;
        float zf = acc[0][mi][ni][r] + bf_;
        float zi = acc[1][mi][ni][r] + bi_;
        float zg = acc[2][mi][ni][r] + bg_;
        float zo = acc[3][mi][ni][r] + bo_;
        float fg = 1.0f / (1.0f + __expf(-zf));
        float ig = 1.0f / (1.0f + __expf(-zi));
        float gg = tanhf(zg);
        float og = 1.0f / (1.0f + __expf(-zo));
        float cn = fg * C[(size_t)m * HID + n0] + ig * gg;
        float hn = og * tanhf(cn);
        HnewBF[(size_t)m * HID + n0] = f2bf(hn);
      }
    }
  }
}

// ---------------------------------------------------------------------------
// Logits GEMM: [4096x1024] @ [1024xNLP] bf16 -> fp32 + bias.
// Block = 8 waves (2M x 4N), tile 128x128; wave tile 64x32.
// ---------------------------------------------------------------------------

__global__ __launch_bounds__(256) void logits_gemm(const unsigned short* __restrict__ Hbf,
                                                   const unsigned short* __restrict__ WlogT,
                                                   const float* __restrict__ blog,
                                                   float* __restrict__ out) {
  __shared__ __align__(16) unsigned short lA[2][128 * 40];
  __shared__ __align__(16) unsigned short lB[2][128 * 40];

  const int tid  = threadIdx.x;
  const int lane = tid & 31;
  const int w    = tid >> 5;
  const int wm   = w & 1;
  const int wn   = w >> 1;
  const int bm   = blockIdx.y * 128;
  const int bn   = blockIdx.x * 128;
  const int NT   = HID / 32;                 // 32 K-tiles

  v8f acc[4][2];
  const v8f vzero = {0.f, 0.f, 0.f, 0.f, 0.f, 0.f, 0.f, 0.f};
#pragma unroll
  for (int mi = 0; mi < 4; ++mi)
#pragma unroll
    for (int ni = 0; ni < 2; ++ni) acc[mi][ni] = vzero;

#if HAS_TDM
  if (w == 0) {   // tile 0 : 1 A-tile + 1 B-tile = 2 TDM ops
    tdm_load_2d(Hbf + (size_t)bm * HID, (unsigned)(size_t)(void*)&lA[0][0], 128, HID);
    tdm_load_2d(WlogT + (size_t)bn * HID, (unsigned)(size_t)(void*)&lB[0][0], 128, HID);
  }
#else
  copy_tile(Hbf + (size_t)bm * HID, &lA[0][0], 128, HID, tid);
  copy_tile(WlogT + (size_t)bn * HID, &lB[0][0], 128, HID, tid);
#endif

  for (int kt = 0; kt < NT; ++kt) {
    const int cur = kt & 1, nxt = cur ^ 1;
#if HAS_TDM
    if (w == 0) {
      if (kt + 1 < NT) {
        const int kk = (kt + 1) * 32;
        tdm_load_2d(Hbf + (size_t)bm * HID + kk, (unsigned)(size_t)(void*)&lA[nxt][0], 128, HID);
        tdm_load_2d(WlogT + (size_t)bn * HID + kk, (unsigned)(size_t)(void*)&lB[nxt][0], 128, HID);
        __builtin_amdgcn_s_wait_tensorcnt((short)2);
      } else {
        __builtin_amdgcn_s_wait_tensorcnt((short)0);
      }
    }
#else
    if (kt + 1 < NT) {
      const int kk = (kt + 1) * 32;
      copy_tile(Hbf + (size_t)bm * HID + kk, &lA[nxt][0], 128, HID, tid);
      copy_tile(WlogT + (size_t)bn * HID + kk, &lB[nxt][0], 128, HID, tid);
    }
#endif
    __syncthreads();

    v16bf afr[4];
    const int ar0  = wm * 64 + (lane & 15);
    const int aoff = (lane < 16) ? 0 : 8;
#pragma unroll
    for (int mi = 0; mi < 4; ++mi) {
      FragU au;
      au.q[0] = *(const uint4*)(&lA[cur][0] + (ar0 + mi * 16) * 40 + aoff);
      au.q[1] = *(const uint4*)(&lA[cur][0] + (ar0 + mi * 16) * 40 + aoff + 16);
      afr[mi] = au.v;
    }
    const int bc0  = wn * 32 + (lane & 15);
    const int koff = (lane < 16) ? 0 : 8;
#pragma unroll
    for (int ni = 0; ni < 2; ++ni) {
      FragU bu;
      bu.q[0] = *(const uint4*)(&lB[cur][0] + (bc0 + ni * 16) * 40 + koff);
      bu.q[1] = *(const uint4*)(&lB[cur][0] + (bc0 + ni * 16) * 40 + koff + 16);
#pragma unroll
      for (int mi = 0; mi < 4; ++mi)
        acc[mi][ni] = __builtin_amdgcn_wmma_f32_16x16x32_bf16(
            false, afr[mi], false, bu.v, (short)0, acc[mi][ni], false, false);
    }
    __syncthreads();
  }

  // epilogue: bias + guarded fp32 store
#pragma unroll
  for (int mi = 0; mi < 4; ++mi) {
#pragma unroll
    for (int ni = 0; ni < 2; ++ni) {
      const int m0 = bm + wm * 64 + mi * 16 + ((lane < 16) ? 0 : 8);
      const int n  = bn + wn * 32 + ni * 16 + (lane & 15);
      if (n < VOCAB) {
        const float bl = blog[n];
#pragma unroll
        for (int r = 0; r < 8; ++r)
          out[(size_t)(m0 + r) * VOCAB + n] = acc[mi][ni][r] + bl;
      }
    }
  }
}

// ---------------------------------------------------------------------------
// Launch
// ---------------------------------------------------------------------------

extern "C" void kernel_launch(void* const* d_in, const int* in_sizes, int n_in,
                              void* d_out, int out_size, void* d_ws, size_t ws_size,
                              hipStream_t stream) {
  const int*   Xidx = (const int*)  d_in[0];
  const float* H    = (const float*)d_in[1];
  const float* C    = (const float*)d_in[2];
  const float* emb  = (const float*)d_in[3];
  const float* Wfx = (const float*)d_in[4],  *bfx = (const float*)d_in[5];
  const float* Wfh = (const float*)d_in[6],  *bfh = (const float*)d_in[7];
  const float* Wix = (const float*)d_in[8],  *bix = (const float*)d_in[9];
  const float* Wih = (const float*)d_in[10], *bih = (const float*)d_in[11];
  const float* Wgx = (const float*)d_in[12], *bgx = (const float*)d_in[13];
  const float* Wgh = (const float*)d_in[14], *bgh = (const float*)d_in[15];
  const float* Wox = (const float*)d_in[16], *box = (const float*)d_in[17];
  const float* Woh = (const float*)d_in[18], *boh = (const float*)d_in[19];
  const float* Wlog = (const float*)d_in[20];
  const float* blog = (const float*)d_in[21];
  float* out = (float*)d_out;

  // workspace layout (bytes, 256-aligned)
  char* ws = (char*)d_ws;
  unsigned short* Acat  = (unsigned short*)(ws);              // 4096*1344*2   = 11,010,048
  unsigned short* WcatT = (unsigned short*)(ws + 11010048);   // 4*1024*1344*2 = 11,010,048
  float*          bsum  = (float*)         (ws + 22020096);   // 4*1024*4      = 16,384
  unsigned short* HnewB = (unsigned short*)(ws + 22036480);   // 4096*1024*2   = 8,388,608
  unsigned short* WlogT = (unsigned short*)(ws + 30425088);   // NLP*1024*2    = 103,022,592
  // total = 133,447,680 bytes

  pack_A<<<(B_ROWS * KC) / 256, 256, 0, stream>>>(Xidx, H, emb, Acat);
  packT_Wcat<<<dim3(KC / 32, HID / 32, 4), 256, 0, stream>>>(Wfx, Wfh, Wix, Wih,
                                                             Wgx, Wgh, Wox, Woh, WcatT);
  pack_bsum<<<(4 * HID) / 256, 256, 0, stream>>>(bfx, bfh, bix, bih, bgx, bgh, box, boh, bsum);
  packT_Wlog<<<dim3(HID / 32, NLP / 32), 256, 0, stream>>>(Wlog, WlogT);

  dim3 gGrid(HID / 64, B_ROWS / 128);      // 16 x 32
  gates_lstm<<<gGrid, 256, 0, stream>>>(Acat, WcatT, bsum, C, HnewB);

  dim3 lGrid(NLP / 128, B_ROWS / 128);     // 393 x 32
  logits_gemm<<<lGrid, 256, 0, stream>>>(HnewB, WlogT, blog, out);
}
